// MultiLevelSpikeDrivenSelfAttention_57552561766460
// MI455X (gfx1250) — matrix operand order repaired and checked by reading, not compiled, not run on hardware
//
#include <hip/hip_runtime.h>
#include <hip/hip_bf16.h>
#include <math.h>

// ---------------------------------------------------------------------------
// MultiLevelSpikeDrivenSelfAttention for MI455X (gfx1250, wave32, WMMA + TDM)
//   D=1024, H=16, DH=64, B=2, T=1920, scales {1,3,5}
// GEMMs run through v_wmma_f32_16x16x32_f16 (fp16 in, fp32 accumulate).
// Attention q/k tiles are staged global->LDS by the Tensor Data Mover
// (tensor_load_to_lds + s_wait_tensorcnt), with D# hardware padding used to
// produce the 72-half padded LDS rows the WMMA fragment loaders expect.
// ---------------------------------------------------------------------------

typedef __attribute__((ext_vector_type(16))) _Float16 v16h;
typedef __attribute__((ext_vector_type(8)))  _Float16 v8h;
typedef __attribute__((ext_vector_type(8)))  float    v8f;
typedef __attribute__((ext_vector_type(4)))  unsigned u32x4;
typedef __attribute__((ext_vector_type(8)))  unsigned u32x8;

#define D_MODEL 1024
#define N_HEAD  16
#define D_HEAD  64
#define BATCH   2
#define SEQ_T   1920
#define TH_SPK  1.0f

#define USE_TDM 1   // Tensor Data Mover staging for attention q/k tiles

__device__ inline float sigf(float x) { return 1.0f / (1.0f + __expf(-x)); }

// ---------------------------------------------------------------------------
// WMMA fragment loaders (CDNA5 16x16x32 f16 layouts, ISA 7.12.2).
// A (16xK tile, row-major (M,K) buffer): lane m=l&15, half hl=l>>4.
//   elements 0..7  = K [8*hl   .. 8*hl+7]    (contiguous 16B)
//   elements 8..15 = K [16+8*hl.. 16+8*hl+7] (contiguous 16B)
// B (KxN tile from an (N,K) row-major buffer): lane col n=l&15.
//   elements 0..7 = K [8*hl .. 8*hl+7], 8..15 = K [16+8*hl .. 16+8*hl+7]
// ---------------------------------------------------------------------------
__device__ inline v16h combine16(v8h lo, v8h hi) {
    return __builtin_shufflevector(lo, hi, 0, 1, 2, 3, 4, 5, 6, 7,
                                   8, 9, 10, 11, 12, 13, 14, 15);
}

__device__ inline v16h load_frag_a(const _Float16* lds, int ld, int row0, int k0) {
    int lane = threadIdx.x & 31;
    int hl   = lane >> 4;
    int m    = lane & 15;
    const _Float16* base = lds + (row0 + m) * ld + k0;
    v8h lo = *(const v8h*)(base + 8 * hl);
    v8h hi = *(const v8h*)(base + 16 + 8 * hl);
    return combine16(lo, hi);
}

__device__ inline v16h load_frag_b_nk(const _Float16* lds, int ld, int col0, int k0) {
    int lane = threadIdx.x & 31;
    int hl   = lane >> 4;
    int n    = lane & 15;
    const _Float16* base = lds + (col0 + n) * ld + k0;
    v8h lo = *(const v8h*)(base + 8 * hl);
    v8h hi = *(const v8h*)(base + 16 + 8 * hl);
    return combine16(lo, hi);
}

__device__ inline v8f wmma_f16(v16h a, v16h b, v8f c) {
    return __builtin_amdgcn_wmma_f32_16x16x32_f16(false, a, false, b,
                                                  (short)0, c, false, false);
}

// ---------------------------------------------------------------------------
// Tensor Data Mover: 2D f16 tile load global->LDS (ISA 08_async_tensor §8).
// tile/tensor dims = rows x cols (cols contiguous), row stride in elements.
// LDS rows are padded 64->72 halves via D# pad (interval 32 DWORDs, pad 4).
// ---------------------------------------------------------------------------
#if USE_TDM
__device__ inline unsigned lds_addr_of(const void* p) {
    return (unsigned)(unsigned long long)
        (__attribute__((address_space(3))) const char*)p;
}

__device__ inline void tdm_load_tile_2d(unsigned lds_byte_addr,
                                        const _Float16* gptr,
                                        unsigned rows, unsigned cols,
                                        unsigned row_stride_elems) {
    unsigned long long ga = (unsigned long long)(size_t)gptr;
    u32x4 g0;
    g0[0] = 1u;                                        // count=1, user mode
    g0[1] = lds_byte_addr;                             // lds_addr
    g0[2] = (unsigned)(ga & 0xFFFFFFFFu);              // global_addr[31:0]
    g0[3] = (unsigned)((ga >> 32) & 0x1FFFFFFu)        // global_addr[56:32]
            | (2u << 30);                              // type = 2 (image)
    u32x8 g1;
    g1[0] = (1u << 16)                                 // data_size = 2 bytes
            | (1u << 20)                               // pad_enable
            | (4u << 22)                               // pad_interval: 32 DWORDs
            | (3u << 25);                              // pad_amount: 4 DWORDs
    g1[1] = (cols & 0xFFFFu) << 16;                    // tensor_dim0[15:0]
    g1[2] = ((cols >> 16) & 0xFFFFu)                   // tensor_dim0[31:16]
            | ((rows & 0xFFFFu) << 16);                // tensor_dim1[15:0]
    g1[3] = ((rows >> 16) & 0xFFFFu)                   // tensor_dim1[31:16]
            | ((cols & 0xFFFFu) << 16);                // tile_dim0
    g1[4] = rows & 0xFFFFu;                            // tile_dim1 (tile_dim2=0)
    g1[5] = row_stride_elems;                          // tensor_dim0_stride lo
    g1[6] = 0u;                                        // stride0 hi | stride1 lo
    g1[7] = 0u;                                        // stride1 hi
    asm volatile("tensor_load_to_lds %0, %1" :: "s"(g0), "s"(g1) : "memory");
}

__device__ inline void wait_tensorcnt0() {
#if __has_builtin(__builtin_amdgcn_s_wait_tensorcnt)
    __builtin_amdgcn_s_wait_tensorcnt(0);
#else
    asm volatile("s_wait_tensorcnt 0x0" ::: "memory");
#endif
}
#endif // USE_TDM

// ---------------------------------------------------------------------------
// fp32 -> fp16 conversion (grid-stride)
// ---------------------------------------------------------------------------
__global__ void cvt_f32_f16(const float* __restrict__ in,
                            _Float16* __restrict__ out, int n) {
    for (int i = blockIdx.x * blockDim.x + threadIdx.x; i < n;
         i += gridDim.x * blockDim.x)
        out[i] = (_Float16)in[i];
}

// ---------------------------------------------------------------------------
// fp32 (K,N) -> fp16 (N,K) transposed weights, LDS-tiled 32x32.
// ---------------------------------------------------------------------------
__global__ __launch_bounds__(256)
void transpose_cvt(const float* __restrict__ in, _Float16* __restrict__ out,
                   int K, int N) {
    __shared__ float tile[32][33];
    int nx = blockIdx.x * 32 + threadIdx.x;
    int ky = blockIdx.y * 32 + threadIdx.y;
#pragma unroll
    for (int j = 0; j < 32; j += 8)
        tile[threadIdx.y + j][threadIdx.x] = in[(size_t)(ky + j) * N + nx];
    __syncthreads();
    int kx = blockIdx.y * 32 + threadIdx.x;
    int ny = blockIdx.x * 32 + threadIdx.y;
#pragma unroll
    for (int j = 0; j < 32; j += 8)
        out[(size_t)(ny + j) * K + kx] = (_Float16)tile[threadIdx.x][threadIdx.y + j];
}

// ---------------------------------------------------------------------------
// Generic WMMA GEMM:  C(MxN) = A(MxK) @ W(KxN) + bias, fused epilogue.
// A row-major (M,K) f16; Wt TRANSPOSED (N,K) f16.
// Block tile 128x128, 256 threads = 8 waves; wave = 16x128 strip (8 WMMA/step).
// All 8 B fragments are loaded before the WMMA burst so the scheduler can
// overlap ds_load_b128 latency with the matrix pipe.
// mode 0: spike + sparsity gate -> f16 | mode 1: plain f16 | mode 2: spike f32
// ---------------------------------------------------------------------------
#define G_LD 40   // 32 + 8 pad halves (16B-aligned rows)
__global__ __launch_bounds__(256)
void gemm_wmma_f16_k(const _Float16* __restrict__ A,
                     const _Float16* __restrict__ Wt,
                     const float* __restrict__ bias,
                     void* __restrict__ Cout,
                     const float* __restrict__ thrp,
                     int M, int N, int K, int mode) {
    __shared__ _Float16 As[128 * G_LD];   // (M,K) tile
    __shared__ _Float16 Bs[128 * G_LD];   // (N,K) tile

    const int tid  = threadIdx.x;
    const int lane = tid & 31;
    const int wv   = tid >> 5;
    const int m0   = wv * 16;
    const int row0 = blockIdx.y * 128;
    const int col0 = blockIdx.x * 128;

    float th   = *thrp;
    float gate = sigf(1.0f - th);

    v8f acc[8];
#pragma unroll
    for (int nt = 0; nt < 8; ++nt)
#pragma unroll
        for (int e = 0; e < 8; ++e) acc[nt][e] = 0.0f;

    const int sr0 = tid >> 2;
    const int sc0 = (tid & 3) * 8;

    for (int kk = 0; kk < K; kk += 32) {
        __syncthreads();
#pragma unroll
        for (int j = 0; j < 2; ++j) {
            int r = sr0 + j * 64;
            *(v8h*)&As[r * G_LD + sc0] =
                *(const v8h*)&A[(size_t)(row0 + r) * K + kk + sc0];
            *(v8h*)&Bs[r * G_LD + sc0] =
                *(const v8h*)&Wt[(size_t)(col0 + r) * K + kk + sc0];
        }
        if (kk + 32 < K) {   // next-tile prefetch -> global_prefetch_b8
            __builtin_prefetch(&A[(size_t)(row0 + sr0) * K + kk + 32 + sc0], 0, 1);
            __builtin_prefetch(&Wt[(size_t)(col0 + sr0) * K + kk + 32 + sc0], 0, 1);
        }
        __syncthreads();

        v16h a = load_frag_a(As, G_LD, m0, 0);
        v16h bf[8];
#pragma unroll
        for (int nt = 0; nt < 8; ++nt)
            bf[nt] = load_frag_b_nk(Bs, G_LD, nt * 16, 0);
#pragma unroll
        for (int nt = 0; nt < 8; ++nt)
            acc[nt] = wmma_f16(a, bf[nt], acc[nt]);
    }

    const int mlb = 8 * (lane >> 4);
    const int nl  = lane & 15;
#pragma unroll
    for (int nt = 0; nt < 8; ++nt) {
#pragma unroll
        for (int r = 0; r < 8; ++r) {
            int grow = row0 + m0 + r + mlb;
            int gcol = col0 + nt * 16 + nl;
            float c  = acc[nt][r] + bias[gcol];
            size_t o = (size_t)grow * N + gcol;
            if (mode == 0) {
                ((_Float16*)Cout)[o] = (_Float16)((c > TH_SPK) ? gate : 0.0f);
            } else if (mode == 1) {
                ((_Float16*)Cout)[o] = (_Float16)c;
            } else {
                ((float*)Cout)[o] = (c > TH_SPK) ? 1.0f : 0.0f;
            }
        }
    }
}

// ---------------------------------------------------------------------------
// avg_pool1d along time: (B,T,C) -> (B,T/s,C)
// ---------------------------------------------------------------------------
__global__ void pool_avg(const _Float16* __restrict__ in,
                         _Float16* __restrict__ out,
                         int Bc, int T, int s, int C) {
    int Ts = T / s;
    int total = Bc * Ts * C;
    float inv = 1.0f / (float)s;
    for (int i = blockIdx.x * blockDim.x + threadIdx.x; i < total;
         i += gridDim.x * blockDim.x) {
        int c  = i % C;
        int ts = (i / C) % Ts;
        int b  = i / (C * Ts);
        float sum = 0.0f;
        for (int j = 0; j < s; ++j)
            sum += (float)in[((size_t)b * T + ts * s + j) * C + c];
        out[i] = (_Float16)(sum * inv);
    }
}

// ---------------------------------------------------------------------------
// per-(row, head) squared norms
// ---------------------------------------------------------------------------
__global__ void head_norms(const _Float16* __restrict__ t,
                           float* __restrict__ nrm, int rows) {
    int gid = blockIdx.x * blockDim.x + threadIdx.x;
    if (gid >= rows * N_HEAD) return;
    int row = gid >> 4;
    int h   = gid & 15;
    const _Float16* p = t + (size_t)row * D_MODEL + h * D_HEAD;
    float s = 0.0f;
#pragma unroll
    for (int d = 0; d < D_HEAD; d += 8) {
        v8h v = *(const v8h*)(p + d);
#pragma unroll
        for (int e = 0; e < 8; ++e) {
            float x = (float)v[e];
            s += x * x;
        }
    }
    nrm[gid] = s;
}

// ---------------------------------------------------------------------------
// XNOR-sigmoid attention for one scale, WMMA end to end.
//   scores = sigmoid(64 - |q|^2 - |k|^2 + 2 q.k) ;  O = scores @ V
// Block: 128 threads = 4 waves; 64 q-rows x 64 (=DH) output tile per block.
// q/k tiles staged by TDM with hardware row padding; V staged transposed
// (d,key) by VGPR scatter so the w@V B operand is (N,K)-major.
// ---------------------------------------------------------------------------
#define AT_LD 72  // 64 + 8 pad halves (matches TDM pad: 32 DWORDs + 4 DWORDs)
__global__ __launch_bounds__(128)
void attn_wmma_k(const _Float16* __restrict__ q,
                 const _Float16* __restrict__ k,
                 const _Float16* __restrict__ v,
                 const float* __restrict__ qn,
                 const float* __restrict__ kn,
                 _Float16* __restrict__ cat,
                 int Ts, int s, int scale_idx, int T) {
    __shared__ _Float16 qs_s[64 * AT_LD];  // (qrow, d)
    __shared__ _Float16 ks_s[64 * AT_LD];  // (key, d)  == (N,K) for scores
    __shared__ _Float16 vt_s[64 * AT_LD];  // (d, key)  == (N,K) for w@V
    __shared__ _Float16 ws_s[64 * AT_LD];  // (qrow, key) A operand for w@V
    __shared__ float qn_s[64];
    __shared__ float kn_s[64];

    const int tid  = threadIdx.x;
    const int lane = tid & 31;
    const int wv   = tid >> 5;
    const int m0   = wv * 16;
    const int b    = blockIdx.z;
    const int h    = blockIdx.y;
    const int qr0  = blockIdx.x * 64;

    // ---- stage q tile ----
#if USE_TDM
    if (wv == 0) {
        tdm_load_tile_2d(lds_addr_of(qs_s),
                         &q[(size_t)(b * Ts + qr0) * D_MODEL + h * D_HEAD],
                         64u, 64u, (unsigned)D_MODEL);
    }
#else
#pragma unroll
    for (int j = 0; j < 4; ++j) {
        int seg = tid + j * 128;
        int r   = seg >> 3;
        int c8  = (seg & 7) * 8;
        *(v8h*)&qs_s[r * AT_LD + c8] =
            *(const v8h*)&q[(size_t)(b * Ts + qr0 + r) * D_MODEL + h * D_HEAD + c8];
    }
#endif
    if (tid < 64) qn_s[tid] = qn[(size_t)(b * Ts + qr0 + tid) * N_HEAD + h];

    v8f oacc[4];
#pragma unroll
    for (int nt = 0; nt < 4; ++nt)
#pragma unroll
        for (int e = 0; e < 8; ++e) oacc[nt][e] = 0.0f;

    const int mlb = 8 * (lane >> 4);
    const int nl  = lane & 15;
    const int nChunks = Ts / 64;

    for (int kc = 0; kc < nChunks; ++kc) {
        __syncthreads();   // all waves done reading previous k/v/w tiles
#if USE_TDM
        if (wv == 0) {
            tdm_load_tile_2d(lds_addr_of(ks_s),
                             &k[(size_t)(b * Ts + kc * 64) * D_MODEL + h * D_HEAD],
                             64u, 64u, (unsigned)D_MODEL);
        }
#endif
#pragma unroll
        for (int j = 0; j < 4; ++j) {
            int seg = tid + j * 128;
            int r   = seg >> 3;          // key row within chunk
            int c8  = (seg & 7) * 8;     // d offset
            size_t grow = (size_t)(b * Ts + kc * 64 + r) * D_MODEL + h * D_HEAD + c8;
#if !USE_TDM
            *(v8h*)&ks_s[r * AT_LD + c8] = *(const v8h*)&k[grow];
#endif
            v8h vv = *(const v8h*)&v[grow];
#pragma unroll
            for (int e = 0; e < 8; ++e)          // transposed scatter (d,key)
                vt_s[(c8 + e) * AT_LD + r] = vv[e];
        }
        if (tid < 64) kn_s[tid] = kn[(size_t)(b * Ts + kc * 64 + tid) * N_HEAD + h];
#if USE_TDM
        if (wv == 0) wait_tensorcnt0();   // q (first iter) + k tile landed
#endif
        __syncthreads();

        // ---- scores: S = q @ k^T (16x64 strip per wave) ----
#pragma unroll
        for (int nt = 0; nt < 4; ++nt) {
            v8f sc;
#pragma unroll
            for (int e = 0; e < 8; ++e) sc[e] = 0.0f;
            sc = wmma_f16(load_frag_a(qs_s, AT_LD, m0, 0),
                          load_frag_b_nk(ks_s, AT_LD, nt * 16, 0), sc);
            sc = wmma_f16(load_frag_a(qs_s, AT_LD, m0, 32),
                          load_frag_b_nk(ks_s, AT_LD, nt * 16, 32), sc);
#pragma unroll
            for (int r = 0; r < 8; ++r) {
                int ml = m0 + r + mlb;
                float scv = (float)D_HEAD - qn_s[ml] - kn_s[nt * 16 + nl]
                            + 2.0f * sc[r];
                ws_s[ml * AT_LD + nt * 16 + nl] = (_Float16)sigf(scv);
            }
        }
        __syncthreads();

        // ---- O += W @ V ----
#pragma unroll
        for (int nt = 0; nt < 4; ++nt) {
            oacc[nt] = wmma_f16(load_frag_a(ws_s, AT_LD, m0, 0),
                                load_frag_b_nk(vt_s, AT_LD, nt * 16, 0), oacc[nt]);
            oacc[nt] = wmma_f16(load_frag_a(ws_s, AT_LD, m0, 32),
                                load_frag_b_nk(vt_s, AT_LD, nt * 16, 32), oacc[nt]);
        }
    }

    // ---- epilogue: upsample (repeat s) into cat[b, t, scale_idx*1024 + h*64 + d]
#pragma unroll
    for (int nt = 0; nt < 4; ++nt) {
#pragma unroll
        for (int r = 0; r < 8; ++r) {
            int ml  = m0 + r + mlb;
            int tq  = qr0 + ml;
            int col = scale_idx * D_MODEL + h * D_HEAD + nt * 16 + nl;
            _Float16 val = (_Float16)oacc[nt][r];
            for (int j = 0; j < s; ++j)
                cat[((size_t)b * T + tq * s + j) * (3 * D_MODEL) + col] = val;
        }
    }
}

// ---------------------------------------------------------------------------
// Host-side launch
// ---------------------------------------------------------------------------
extern "C" void kernel_launch(void* const* d_in, const int* in_sizes, int n_in,
                              void* d_out, int out_size, void* d_ws, size_t ws_size,
                              hipStream_t stream) {
    (void)in_sizes; (void)n_in; (void)out_size; (void)ws_size;

    const float* x   = (const float*)d_in[0];
    const float* Wq  = (const float*)d_in[1];
    const float* bq  = (const float*)d_in[2];
    const float* Wk  = (const float*)d_in[3];
    const float* bk  = (const float*)d_in[4];
    const float* Wv  = (const float*)d_in[5];
    const float* bv  = (const float*)d_in[6];
    const float* Wo  = (const float*)d_in[7];
    const float* bo  = (const float*)d_in[8];
    const float* thr = (const float*)d_in[9];
    float* out       = (float*)d_out;

    const int BT = BATCH * SEQ_T;
    const int T3 = SEQ_T / 3, T5 = SEQ_T / 5;

    char* base = (char*)d_ws;
    size_t off = 0;
    auto alloc = [&](size_t bytes) -> void* {
        void* p = base + off;
        off += (bytes + 255) & ~(size_t)255;
        return p;
    };
    _Float16* xh  = (_Float16*)alloc((size_t)BT * D_MODEL * 2);
    _Float16* wqt = (_Float16*)alloc((size_t)D_MODEL * D_MODEL * 2);  // (N,K)
    _Float16* wkt = (_Float16*)alloc((size_t)D_MODEL * D_MODEL * 2);
    _Float16* wvt = (_Float16*)alloc((size_t)D_MODEL * D_MODEL * 2);
    _Float16* wot = (_Float16*)alloc((size_t)3 * D_MODEL * D_MODEL * 2);
    _Float16* qg  = (_Float16*)alloc((size_t)BT * D_MODEL * 2);
    _Float16* kg  = (_Float16*)alloc((size_t)BT * D_MODEL * 2);
    _Float16* vh  = (_Float16*)alloc((size_t)BT * D_MODEL * 2);
    _Float16* q3  = (_Float16*)alloc((size_t)BATCH * T3 * D_MODEL * 2);
    _Float16* k3  = (_Float16*)alloc((size_t)BATCH * T3 * D_MODEL * 2);
    _Float16* v3  = (_Float16*)alloc((size_t)BATCH * T3 * D_MODEL * 2);
    _Float16* q5  = (_Float16*)alloc((size_t)BATCH * T5 * D_MODEL * 2);
    _Float16* k5  = (_Float16*)alloc((size_t)BATCH * T5 * D_MODEL * 2);
    _Float16* v5  = (_Float16*)alloc((size_t)BATCH * T5 * D_MODEL * 2);
    float* qn1 = (float*)alloc((size_t)BT * N_HEAD * 4);
    float* kn1 = (float*)alloc((size_t)BT * N_HEAD * 4);
    float* qn3 = (float*)alloc((size_t)BATCH * T3 * N_HEAD * 4);
    float* kn3 = (float*)alloc((size_t)BATCH * T3 * N_HEAD * 4);
    float* qn5 = (float*)alloc((size_t)BATCH * T5 * N_HEAD * 4);
    float* kn5 = (float*)alloc((size_t)BATCH * T5 * N_HEAD * 4);
    _Float16* cat = (_Float16*)alloc((size_t)BT * 3 * D_MODEL * 2);

    // 1) fp32 -> fp16 staging; weights transposed to (N,K)
    {
        int n = BT * D_MODEL;
        cvt_f32_f16<<<(n + 2047) / 2048, 256, 0, stream>>>(x, xh, n);
    }
    dim3 tb(32, 8);
    transpose_cvt<<<dim3(D_MODEL / 32, D_MODEL / 32), tb, 0, stream>>>(Wq, wqt, D_MODEL, D_MODEL);
    transpose_cvt<<<dim3(D_MODEL / 32, D_MODEL / 32), tb, 0, stream>>>(Wk, wkt, D_MODEL, D_MODEL);
    transpose_cvt<<<dim3(D_MODEL / 32, D_MODEL / 32), tb, 0, stream>>>(Wv, wvt, D_MODEL, D_MODEL);
    transpose_cvt<<<dim3(D_MODEL / 32, 3 * D_MODEL / 32), tb, 0, stream>>>(Wo, wot, 3 * D_MODEL, D_MODEL);

    // 2) projections (WMMA)
    dim3 ggrid(D_MODEL / 128, BT / 128);
    gemm_wmma_f16_k<<<ggrid, 256, 0, stream>>>(xh, wqt, bq, qg, thr,
                                               BT, D_MODEL, D_MODEL, 0);
    gemm_wmma_f16_k<<<ggrid, 256, 0, stream>>>(xh, wkt, bk, kg, thr,
                                               BT, D_MODEL, D_MODEL, 0);
    gemm_wmma_f16_k<<<ggrid, 256, 0, stream>>>(xh, wvt, bv, vh, thr,
                                               BT, D_MODEL, D_MODEL, 1);

    // 3) pooling for scales 3 and 5
    auto pool = [&](const _Float16* in, _Float16* o, int s) {
        int n = BATCH * (SEQ_T / s) * D_MODEL;
        pool_avg<<<(n + 255) / 256, 256, 0, stream>>>(in, o, BATCH, SEQ_T, s, D_MODEL);
    };
    pool(qg, q3, 3); pool(kg, k3, 3); pool(vh, v3, 3);
    pool(qg, q5, 5); pool(kg, k5, 5); pool(vh, v5, 5);

    // 4) per-head squared norms
    auto norms = [&](const _Float16* t, float* nrm, int rows) {
        int n = rows * N_HEAD;
        head_norms<<<(n + 255) / 256, 256, 0, stream>>>(t, nrm, rows);
    };
    norms(qg, qn1, BT);          norms(kg, kn1, BT);
    norms(q3, qn3, BATCH * T3);  norms(k3, kn3, BATCH * T3);
    norms(q5, qn5, BATCH * T5);  norms(k5, kn5, BATCH * T5);

    // 5) attention per scale (writes upsampled rows into cat)
    attn_wmma_k<<<dim3(SEQ_T / 64, N_HEAD, BATCH), 128, 0, stream>>>(
        qg, kg, vh, qn1, kn1, cat, SEQ_T, 1, 0, SEQ_T);
    attn_wmma_k<<<dim3(T3 / 64, N_HEAD, BATCH), 128, 0, stream>>>(
        q3, k3, v3, qn3, kn3, cat, T3, 3, 1, SEQ_T);
    attn_wmma_k<<<dim3(T5 / 64, N_HEAD, BATCH), 128, 0, stream>>>(
        q5, k5, v5, qn5, kn5, cat, T5, 5, 2, SEQ_T);

    // 6) output projection + spike -> f32 {0,1}
    gemm_wmma_f16_k<<<dim3(D_MODEL / 128, BT / 128), 256, 0, stream>>>(
        cat, wot, bo, out, thr, BT, D_MODEL, 3 * D_MODEL, 2);
}
